// BoundedParaboloids_4174708211922
// MI455X (gfx1250) — matrix operand order repaired and checked by reading, not compiled, not run on hardware
//
#include <hip/hip_runtime.h>
#include <hip/hip_bf16.h>

// BoundedParaboloids on MI455X (gfx1250):
//   out[b,u] = mult[u] * sigmoid(sharp[u] * (1 - sum_f (x[b,f]+s[u,f])^2 / sa[u,f]^2))
// Decomposed into an augmented [8192x256]x[256x256] FP16 WMMA GEMM (f32 accum)
// plus a pointwise epilogue. ~12 MB mandatory HBM traffic -> bandwidth bound;
// weights (128 KB f16) pre-swizzled into WMMA B-fragment order so each lane's
// B operand is a single contiguous 32-byte load.

typedef __attribute__((ext_vector_type(16))) _Float16 v16h;
typedef __attribute__((ext_vector_type(8)))  float    v8f;

#define BPB 8192   // batch
#define UPB 256    // units
#define FPB 128    // features
#define KTOT 256   // augmented K = 2*F
#define NKT 8      // K tiles of 32
#define NNT 16     // N tiles of 16

// ---------------------------------------------------------------------------
// Stage 1a: pack W into WMMA B-fragment order.
// W[k, u]: k <  128 -> inv[u,k]           (multiplies x^2 term)
//          k >= 128 -> 2*s[u,k-128]*inv   (multiplies x term)
// B-fragment k order chosen here for V_WMMA_F32_16X16X32_F16 (32x16, wave32; A uses the same):
//   lane L holds column N = L%16; element j (0..15) holds K = (L/16)*16 + j.
// Fragment buffer: [kt][nt][lane][16 f16], 8*16*32*16 = 65536 f16 = 128 KB.
// ---------------------------------------------------------------------------
__global__ __launch_bounds__(128) void bp_prep_w(const float* __restrict__ shift,
                                                 const float* __restrict__ sa,
                                                 _Float16* __restrict__ wf) {
    const int t    = blockIdx.x * blockDim.x + threadIdx.x;   // 0..4095
    const int lane = t & 31;
    const int nt   = (t >> 5) & 15;
    const int kt   = t >> 9;
    const int n    = lane & 15;
    const int h    = lane >> 4;
    const int u    = nt * 16 + n;

    v16h w;
#pragma unroll
    for (int j = 0; j < 16; ++j) {
        const int K = kt * 32 + h * 16 + j;      // 0..255
        const int f = K & (FPB - 1);
        const float a   = sa[u * FPB + f];
        const float inv = 1.0f / (a * a);
        const float val = (K < FPB) ? inv : 2.0f * shift[u * FPB + f] * inv;
        w[j] = (_Float16)val;
    }
    *reinterpret_cast<v16h*>(wf + (size_t)t * 16) = w;
}

// ---------------------------------------------------------------------------
// Stage 1b: c[u] = sum_f s^2 * inv  (f32, exact)
// ---------------------------------------------------------------------------
__global__ __launch_bounds__(256) void bp_prep_c(const float* __restrict__ shift,
                                                 const float* __restrict__ sa,
                                                 float* __restrict__ cvec) {
    const int u = blockIdx.x * blockDim.x + threadIdx.x;
    if (u < UPB) {
        float acc = 0.0f;
#pragma unroll 4
        for (int f = 0; f < FPB; ++f) {
            const float s = shift[u * FPB + f];
            const float a = sa[u * FPB + f];
            acc += (s * s) / (a * a);
        }
        cvec[u] = acc;
    }
}

// ---------------------------------------------------------------------------
// Stage 2: GEMM + epilogue.
// Block = 128 threads (4 waves). Block owns rows [16*blockIdx.x, +16).
// Wave w owns columns [64w, 64w+64): 4 accumulator tiles, A fragment reused 4x.
// A fragment k order used here (16x32 f16, wave32): lane L holds row M = L%16;
//   h = L/16; element j (0..15) -> K = h*16 + j (matches B fragment order).
// A[b, k] = k<128 ? x[b,k]^2 : x[b,k-128]; built from float2 loads of x.
// ---------------------------------------------------------------------------
__global__ __launch_bounds__(128) void bp_gemm(const float*    __restrict__ x,
                                               const _Float16* __restrict__ wf,
                                               const float*    __restrict__ cvec,
                                               const float*    __restrict__ sharp,
                                               const float*    __restrict__ mult,
                                               float*          __restrict__ out) {
    const int lane = threadIdx.x & 31;
    const int wave = threadIdx.x >> 5;
    const int m0   = blockIdx.x * 16;
    const int mrow = lane & 15;
    const int h    = lane >> 4;

    const float* __restrict__ xrow = x + (size_t)(m0 + mrow) * FPB;

    v8f acc[4];
#pragma unroll
    for (int q = 0; q < 4; ++q) acc[q] = (v8f){};

#pragma unroll
    for (int kt = 0; kt < NKT; ++kt) {
        // ---- build A fragment from x (square for the first half of K) ----
        v16h afrag;
        const int kbase = kt * 32;
#pragma unroll
        for (int p = 0; p < 8; ++p) {
            const int i  = 2 * p;                          // element pair start
            const int K  = h * 16 + i;
            const int kk = kbase + K;                      // augmented k, even
            const int col = kk & (FPB - 1);                // x column (even)
            const float2 xv = *reinterpret_cast<const float2*>(xrow + col);
            float a0, a1;
            if (kk < FPB) { a0 = xv.x * xv.x; a1 = xv.y * xv.y; }
            else          { a0 = xv.x;        a1 = xv.y;        }
            afrag[i]     = (_Float16)a0;
            afrag[i + 1] = (_Float16)a1;
        }
        // ---- 4 N-tiles: contiguous 32B B-fragment load + WMMA ----
#pragma unroll
        for (int q = 0; q < 4; ++q) {
            const int nt = wave * 4 + q;
            const v16h bfrag = *reinterpret_cast<const v16h*>(
                wf + ((size_t)((kt * NNT + nt) * 32 + lane)) * 16);
            acc[q] = __builtin_amdgcn_wmma_f32_16x16x32_f16(
                /*neg_a=*/false, afrag, /*neg_b=*/false, bfrag,
                /*c_mod=*/(short)0, acc[q], /*reuse_a=*/false, /*reuse_b=*/false);
        }
    }

    // ---- epilogue: quad -> sigmoid -> scale; C/D layout: VGPR v holds
    //      M = v + 8*h, N = lane%16 ----
#pragma unroll
    for (int q = 0; q < 4; ++q) {
        const int u  = (wave * 4 + q) * 16 + (lane & 15);
        const float cu = cvec[u];
        const float sh = sharp[u];
        const float mu = mult[u];
#pragma unroll
        for (int v = 0; v < 8; ++v) {
            const int row  = m0 + v + h * 8;
            const float quad = acc[q][v] + cu;
            const float z    = sh * (1.0f - quad);
            const float sg   = 1.0f / (1.0f + __expf(-z));
            out[(size_t)row * UPB + u] = mu * sg;
        }
    }
}

// ---------------------------------------------------------------------------
extern "C" void kernel_launch(void* const* d_in, const int* in_sizes, int n_in,
                              void* d_out, int out_size, void* d_ws, size_t ws_size,
                              hipStream_t stream) {
    const float* x          = (const float*)d_in[0];   // [8192,128]
    const float* shift      = (const float*)d_in[1];   // [256*128]
    const float* semi_axis  = (const float*)d_in[2];   // [256,128]
    const float* sharpness  = (const float*)d_in[3];   // [256]
    const float* multiplier = (const float*)d_in[4];   // [256]
    float* out = (float*)d_out;                        // [8192,256]

    // workspace layout: [0,128KB) f16 W fragments; [128KB, +1KB) c[u]
    _Float16* wfrag = (_Float16*)d_ws;
    float*    cvec  = (float*)((char*)d_ws + (size_t)NKT * NNT * 32 * 16 * sizeof(_Float16));

    bp_prep_w<<<dim3((NKT * NNT * 32) / 128), dim3(128), 0, stream>>>(shift, semi_axis, wfrag);
    bp_prep_c<<<dim3(1), dim3(256), 0, stream>>>(shift, semi_axis, cvec);
    bp_gemm<<<dim3(BPB / 16), dim3(128), 0, stream>>>(x, wfrag, cvec, sharpness, multiplier, out);
}
